// EncoderLayer_60979945669136
// MI455X (gfx1250) — matrix operand order repaired and checked by reading, not compiled, not run on hardware
//
#include <hip/hip_runtime.h>
#include <hip/hip_bf16.h>

typedef __bf16 bf16_t;
typedef __attribute__((ext_vector_type(16))) __bf16 v16bf;
typedef __attribute__((ext_vector_type(8)))  float  v8f;

static constexpr int B_ = 4, S_ = 1024, D_ = 1024, H_ = 16, F_ = 4096, DK_ = 64;

// ---------------------------------------------------------------------------
// Fragment loaders per CDNA5 ISA 7.12.2 layouts (wave32).
// A (16x32 bf16, row-major, MxK): lane L holds row M=L%16; elems 0..7 = K
// kbase..kbase+7, elems 8..15 = K kbase+16..+23, kbase = k0 + 8*(L>=16).
// ---------------------------------------------------------------------------
__device__ __forceinline__ v16bf load_a_frag(const bf16_t* __restrict__ A, int lda,
                                             int m, int k0, int lane) {
    const bf16_t* p = A + (size_t)m * lda + k0 + ((lane >> 4) << 3);
    union { v16bf v; uint4 q[2]; } u;
    u.q[0] = *reinterpret_cast<const uint4*>(p);
    u.q[1] = *reinterpret_cast<const uint4*>(p + 16);
    return u.v;
}

// B (32x16 bf16, KxN), sourced from Bt stored N-major (Bt[n][k], K contiguous):
// lane L holds col N=L%16; elems 0..7 = K kbase..+7, elems 8..15 = K kbase+16..+23, kbase = k0 + 8*(L>=16).
__device__ __forceinline__ v16bf load_b_frag(const bf16_t* __restrict__ Bt, int ldb,
                                             int n, int k0, int lane) {
    const bf16_t* p = Bt + (size_t)n * ldb + k0 + ((lane >> 4) << 3);
    union { v16bf v; uint4 q[2]; } u;
    u.q[0] = *reinterpret_cast<const uint4*>(p);
    u.q[1] = *reinterpret_cast<const uint4*>(p + 16);
    return u.v;
}

// ---------------------------------------------------------------------------
// Generic batched WMMA GEMM: C[M,N] = A[M,K] * Bt[N,K]^T, f32 accumulate.
// Wave tile = (WM*16) x (WN*16); block = WAVES_M x WAVES_N waves.
// Epilogue functor receives (batch, m, n, value).
// ---------------------------------------------------------------------------
template <int WAVES_M, int WAVES_N, int WM, int WN, typename Epi>
__launch_bounds__(WAVES_M* WAVES_N * 32)
__global__ void gemm_wmma(const bf16_t* __restrict__ A, int lda, long long strideA,
                          const bf16_t* __restrict__ Bt, int ldb, long long strideB,
                          int K, Epi epi) {
    const int lane = threadIdx.x & 31;
    const int wave = threadIdx.x >> 5;
    const int wm = wave % WAVES_M;
    const int wn = wave / WAVES_M;
    const int z = blockIdx.z;
    A  += (size_t)z * strideA;
    Bt += (size_t)z * strideB;
    const int m0 = (blockIdx.x * WAVES_M + wm) * (WM * 16);
    const int n0 = (blockIdx.y * WAVES_N + wn) * (WN * 16);

    v8f acc[WM][WN] = {};
    const int mrow = m0 + (lane & 15);
    const int nrow = n0 + (lane & 15);

    for (int k0 = 0; k0 < K; k0 += 32) {
        v16bf a[WM], b[WN];
#pragma unroll
        for (int i = 0; i < WM; ++i) a[i] = load_a_frag(A, lda, mrow + i * 16, k0, lane);
#pragma unroll
        for (int j = 0; j < WN; ++j) b[j] = load_b_frag(Bt, ldb, nrow + j * 16, k0, lane);
#pragma unroll
        for (int i = 0; i < WM; ++i)
#pragma unroll
            for (int j = 0; j < WN; ++j)
                acc[i][j] = __builtin_amdgcn_wmma_f32_16x16x32_bf16(
                    false, a[i], false, b[j], (short)0, acc[i][j], false, false);
    }

    const int lm = lane & 15, lh = lane >> 4;  // C/D: M = r + 8*(lane>=16), N = lane%16
#pragma unroll
    for (int i = 0; i < WM; ++i)
#pragma unroll
        for (int j = 0; j < WN; ++j)
#pragma unroll
            for (int r = 0; r < 8; ++r)
                epi(z, m0 + i * 16 + lh * 8 + r, n0 + j * 16 + lm, acc[i][j][r]);
}

// ---------------------------------------------------------------------------
// Epilogues
// ---------------------------------------------------------------------------
struct EpiQKV {  // N = 3*D: [0,1024)=Q, [1024,2048)=K, [2048,3072)=V(transposed)
    bf16_t *q, *k, *vT;
    __device__ void operator()(int, int m, int n, float c) const {
        int sel = n >> 10, hn = n & 1023, h = hn >> 6, dk = hn & 63;
        int b = m >> 10, s = m & 1023;
        bf16_t v = (bf16_t)c;
        if (sel == 0)      q [(((size_t)(h * B_ + b) * S_ + s) << 6) + dk] = v;
        else if (sel == 1) k [(((size_t)(h * B_ + b) * S_ + s) << 6) + dk] = v;
        else               vT[(((size_t)(h * B_ + b) * DK_ + dk) << 10) + s] = v;
    }
};

struct EpiScores {  // scale, mask, raw = (s+prev)/2 -> streamed NT to d_out
    const unsigned char* pad;
    const float* prev;
    float* raw;
    __device__ void operator()(int z, int m, int n, float c) const {
        int b = z & 3;
        float s = c * 0.125f;  // DK^-0.5
        if (pad[b * S_ + m] | pad[b * S_ + n]) s = -__builtin_inff();
        size_t off = ((size_t)z << 20) + ((size_t)m << 10) + n;
        float p = __builtin_nontemporal_load(prev + off);
        __builtin_nontemporal_store(0.5f * (s + p), raw + off);
    }
};

struct EpiAtt {  // (h,b,s,dk) -> att[b*S+s][h*64+dk], bf16 for proj GEMM
    bf16_t* att;
    __device__ void operator()(int z, int m, int n, float c) const {
        int b = z & 3, h = z >> 2;
        att[(((size_t)(b * S_ + m)) << 10) + h * DK_ + n] = (bf16_t)c;
    }
};

struct EpiProj {  // attention = c + src (residual), fp32
    const float* src;
    float* attn;
    __device__ void operator()(int, int m, int n, float c) const {
        size_t i = ((size_t)m << 10) + n;
        attn[i] = c + src[i];
    }
};

struct EpiFFN1 {  // relu(c + b1) -> bf16
    const float* b1;
    bf16_t* h;
    __device__ void operator()(int, int m, int n, float c) const {
        h[((size_t)m << 12) + n] = (bf16_t)fmaxf(c + b1[n], 0.0f);
    }
};

struct EpiFFN2 {  // out = attention + c + b2, fp32 to d_out
    const float* b2;
    const float* attn;
    float* out;
    __device__ void operator()(int, int m, int n, float c) const {
        size_t i = ((size_t)m << 10) + n;
        out[i] = attn[i] + c + b2[n];
    }
};

// ---------------------------------------------------------------------------
// Elementwise / reduction kernels
// ---------------------------------------------------------------------------
__global__ void rmsnorm_kernel(const float* __restrict__ x, const float* __restrict__ gamma,
                               bf16_t* __restrict__ out) {
    __shared__ float red[256];
    const int row = blockIdx.x, tid = threadIdx.x;
    const float* xr = x + (size_t)row * D_;
    float ss = 0.0f;
    for (int d = tid; d < D_; d += 256) { float v = xr[d]; ss += v * v; }
    red[tid] = ss;
    __syncthreads();
    for (int s = 128; s > 0; s >>= 1) {
        if (tid < s) red[tid] += red[tid + s];
        __syncthreads();
    }
    float rinv = rsqrtf(red[0] * (1.0f / D_));
    for (int d = tid; d < D_; d += 256)
        out[(size_t)row * D_ + d] = (bf16_t)(gamma[d] * xr[d] * rinv);
}

__global__ void softmax_kernel(const float* __restrict__ raw, const int* __restrict__ layer_ind,
                               bf16_t* __restrict__ w) {
    __shared__ float red[256];
    const int tid = threadIdx.x;
    const size_t row = blockIdx.x;
    const float* r = raw + (row << 10);
    const float inv = 1.0f / (1.0f - exp2f(-(float)layer_ind[0]));  // cor = raw/(1-0.5^L)
    float vals[4];
    float lmax = -__builtin_inff();
#pragma unroll
    for (int i = 0; i < 4; ++i) {
        vals[i] = r[tid + i * 256] * inv;
        lmax = fmaxf(lmax, vals[i]);
    }
    red[tid] = lmax;
    __syncthreads();
    for (int s = 128; s > 0; s >>= 1) {
        if (tid < s) red[tid] = fmaxf(red[tid], red[tid + s]);
        __syncthreads();
    }
    float m = red[0];
    float msafe = (m == -__builtin_inff()) ? 0.0f : m;
    __syncthreads();
    float lsum = 0.0f;
#pragma unroll
    for (int i = 0; i < 4; ++i) {
        vals[i] = expf(vals[i] - msafe);
        lsum += vals[i];
    }
    red[tid] = lsum;
    __syncthreads();
    for (int s = 128; s > 0; s >>= 1) {
        if (tid < s) red[tid] += red[tid + s];
        __syncthreads();
    }
    float denom = red[0];
    float dinv = (denom == 0.0f) ? 1.0f : 1.0f / denom;
#pragma unroll
    for (int i = 0; i < 4; ++i)
        w[(row << 10) + tid + i * 256] = (bf16_t)(vals[i] * dinv);
}

__global__ void conv_qkv_kernel(const float* __restrict__ Wq, const float* __restrict__ Wk,
                                const float* __restrict__ Wv, bf16_t* __restrict__ out) {
    size_t idx = (size_t)blockIdx.x * 256 + threadIdx.x;  // 3*3072/3... total 3*1024*1024
    int d = idx & 1023;
    int ndk = (int)(idx >> 10);     // 0..3071
    int dk = ndk & 63, h = (ndk >> 6) & 15, sel = ndk >> 10;
    const float* W = (sel == 0) ? Wq : (sel == 1) ? Wk : Wv;
    out[idx] = (bf16_t)W[((size_t)h * D_ + d) * DK_ + dk];  // Wt[n][d] = W[h][d][dk]
}

__global__ void transpose_conv_kernel(const float* __restrict__ in, bf16_t* __restrict__ out,
                                      int R, int C) {
    size_t idx = (size_t)blockIdx.x * 256 + threadIdx.x;  // enumerates output [C][R]
    if (idx >= (size_t)R * C) return;
    int r = (int)(idx % R);
    int c = (int)(idx / R);
    out[idx] = (bf16_t)in[(size_t)r * C + c];
}

// ---------------------------------------------------------------------------
// Host launcher
// ---------------------------------------------------------------------------
extern "C" void kernel_launch(void* const* d_in, const int* /*in_sizes*/, int /*n_in*/,
                              void* d_out, int /*out_size*/, void* d_ws, size_t /*ws_size*/,
                              hipStream_t stream) {
    const float* src        = (const float*)d_in[0];
    const unsigned char* pad= (const unsigned char*)d_in[1];  // bool mask (1B/elem)
    const float* prev       = (const float*)d_in[2];
    const int*   layer_ind  = (const int*)d_in[3];
    const float* Wq         = (const float*)d_in[4];
    const float* Wk         = (const float*)d_in[5];
    const float* Wv         = (const float*)d_in[6];
    const float* Wproj      = (const float*)d_in[7];
    const float* gamma1     = (const float*)d_in[8];
    const float* gamma2     = (const float*)d_in[9];
    const float* W1         = (const float*)d_in[10];
    const float* b1         = (const float*)d_in[11];
    const float* W2         = (const float*)d_in[12];
    const float* b2         = (const float*)d_in[13];

    float* out_f   = (float*)d_out;                          // (B,S,D)
    float* raw_out = out_f + (size_t)B_ * S_ * D_;           // (H,B,S,S)

    char* ws = (char*)d_ws;
    size_t cur = 0;
    auto alloc = [&](size_t bytes) -> void* {
        void* p = ws + cur;
        cur += (bytes + 255) & ~(size_t)255;
        return p;
    };
    const size_t MD  = (size_t)B_ * S_ * D_;   // 4M elems
    const size_t HSK = (size_t)H_ * B_ * S_ * DK_;

    bf16_t* ln1   = (bf16_t*)alloc(MD * 2);
    bf16_t* wqkvt = (bf16_t*)alloc((size_t)3 * D_ * D_ * 2);
    bf16_t* wprjt = (bf16_t*)alloc((size_t)D_ * D_ * 2);
    bf16_t* w1t   = (bf16_t*)alloc((size_t)D_ * F_ * 2);
    bf16_t* w2t   = (bf16_t*)alloc((size_t)D_ * F_ * 2);
    bf16_t* qb    = (bf16_t*)alloc(HSK * 2);
    bf16_t* kb    = (bf16_t*)alloc(HSK * 2);
    bf16_t* vT    = (bf16_t*)alloc(HSK * 2);
    bf16_t* wbuf  = (bf16_t*)alloc((size_t)H_ * B_ * S_ * S_ * 2);  // 128 MB
    bf16_t* attb  = (bf16_t*)alloc(MD * 2);
    float*  attn  = (float*)alloc(MD * 4);
    bf16_t* ln2   = (bf16_t*)alloc(MD * 2);
    bf16_t* hbuf  = (bf16_t*)alloc((size_t)B_ * S_ * F_ * 2);

    // --- weight conversion / transposition to bf16 (K-contiguous B operands)
    conv_qkv_kernel<<<(3 * 1024 * 1024) / 256, 256, 0, stream>>>(Wq, Wk, Wv, wqkvt);
    transpose_conv_kernel<<<(1024 * 1024) / 256, 256, 0, stream>>>(Wproj, wprjt, D_, D_);
    transpose_conv_kernel<<<(1024 * 4096) / 256, 256, 0, stream>>>(W1, w1t, D_, F_);
    transpose_conv_kernel<<<(4096 * 1024) / 256, 256, 0, stream>>>(W2, w2t, F_, D_);

    // --- ln1 = rmsnorm(src, gamma1)
    rmsnorm_kernel<<<B_ * S_, 256, 0, stream>>>(src, gamma1, ln1);

    // --- fused QKV GEMM: (4096 x 3072 x 1024)
    gemm_wmma<2, 2, 4, 4><<<dim3(32, 24, 1), 128, 0, stream>>>(
        ln1, D_, 0LL, wqkvt, D_, 0LL, D_, EpiQKV{qb, kb, vT});

    // --- scores: 64 batches of (1024 x 1024 x 64); fused mask + prev + raw
    gemm_wmma<2, 2, 4, 4><<<dim3(8, 8, H_ * B_), 128, 0, stream>>>(
        qb, DK_, (long long)(S_ * DK_), kb, DK_, (long long)(S_ * DK_), DK_,
        EpiScores{pad, prev, raw_out});

    // --- softmax rows (65536 rows of 1024)
    softmax_kernel<<<H_ * B_ * S_, 256, 0, stream>>>(raw_out, layer_ind, wbuf);

    // --- att = w @ v: 64 batches of (1024 x 64 x 1024); waves tiled 4x1 in M
    gemm_wmma<4, 1, 4, 4><<<dim3(4, 1, H_ * B_), 128, 0, stream>>>(
        wbuf, S_, (long long)((size_t)S_ * S_), vT, S_, (long long)(S_ * DK_), S_,
        EpiAtt{attb});

    // --- projection + residual: (4096 x 1024 x 1024)
    gemm_wmma<2, 2, 4, 4><<<dim3(32, 8, 1), 128, 0, stream>>>(
        attb, D_, 0LL, wprjt, D_, 0LL, D_, EpiProj{src, attn});

    // --- ln2 = rmsnorm(attention, gamma2)
    rmsnorm_kernel<<<B_ * S_, 256, 0, stream>>>(attn, gamma2, ln2);

    // --- FFN1: (4096 x 4096 x 1024), +b1, relu
    gemm_wmma<2, 2, 4, 4><<<dim3(32, 32, 1), 128, 0, stream>>>(
        ln2, D_, 0LL, w1t, D_, 0LL, D_, EpiFFN1{b1, hbuf});

    // --- FFN2: (4096 x 1024 x 4096), +b2 +attention -> out
    gemm_wmma<2, 2, 4, 4><<<dim3(32, 8, 1), 128, 0, stream>>>(
        hbuf, F_, 0LL, w2t, F_, 0LL, F_, EpiFFN2{b2, attn, out_f});
}
